// LSTMPredictor_25726854103409
// MI455X (gfx1250) — matrix operand order repaired and checked by reading, not compiled, not run on hardware
//
#include <hip/hip_runtime.h>

// ---------------- problem constants (match reference) ----------------
#define Bsz     256
#define Lseq    1000
#define Hd      51
#define Fut     100
#define Tt      (Lseq + Fut)   // 1100
#define G4H     204            // 4*H
#define NT      13             // ceil(204/16) gate N-tiles
#define THREADS 256
#define WAVES   8
#define B1_ELEMS (NT*2*512)    // 13312 packed bf16 for layer-1 B fragments
#define B2_ELEMS (NT*4*512)    // 26624 packed bf16 for layer-2 B fragments

typedef __attribute__((ext_vector_type(16))) __bf16       v16bf;
typedef __attribute__((ext_vector_type(8)))  float        v8f;
typedef __attribute__((ext_vector_type(4)))  unsigned int u32x4;

union FragU { v16bf v; u32x4 q[2]; };   // 32B: one WMMA A/B operand per lane

__device__ __forceinline__ v8f wmma_bf16(v16bf a, v16bf b, v8f c) {
  // (neg_a, A, neg_b, B, c_mod, C, reuse_a, reuse_b)
  return __builtin_amdgcn_wmma_f32_16x16x32_bf16(false, a, false, b, (short)0, c, false, false);
}

__device__ __forceinline__ unsigned short f2bf(float f) {
  unsigned u = __float_as_uint(f);
  u += 0x7fffu + ((u >> 16) & 1u);          // RNE
  return (unsigned short)(u >> 16);
}
__device__ __forceinline__ float bf2f(unsigned short h) {
  return __uint_as_float(((unsigned)h) << 16);
}
__device__ __forceinline__ float fsig(float x) {
  return 1.f / (1.f + __expf(-x));
}
__device__ __forceinline__ float ftanh(float x) {
  x = fminf(fmaxf(x, -15.f), 15.f);
  float e = __expf(-2.f * x);
  return (1.f - e) / (1.f + e);
}

// ------------------------------------------------------------------
// Prep: pack W_hh1^T (64x208 padded) and [W_ih2;W_hh2]^T (128x208 padded)
// into bf16 B-fragment order: per (tile,kchunk): 32 lanes x 16 contiguous
// bf16 K-values (lanes 0-15: K 0..15 of chunk; lanes 16-31: K 16..31).
// ------------------------------------------------------------------
__global__ void prep_pack_weights(const float* __restrict__ Whh1,
                                  const float* __restrict__ Wih2,
                                  const float* __restrict__ Whh2,
                                  unsigned short* __restrict__ pack)
{
  int e = blockIdx.x * 256 + threadIdx.x;
  if (e >= B1_ELEMS + B2_ELEMS) return;
  float val = 0.f;
  if (e < B1_ELEMS) {
    int block = e >> 9; int t = block >> 1; int kk = block & 1;
    int r = e & 511;    int lane = r >> 4;  int q = r & 15;
    int N = t * 16 + (lane & 15);
    int K = kk * 32 + ((lane >= 16) ? 16 : 0) + q;
    if (N < G4H && K < Hd) val = Whh1[N * Hd + K];          // B1[k][n] = Whh1[n][k]
  } else {
    int e2 = e - B1_ELEMS;
    int block = e2 >> 9; int t = block >> 2; int kk = block & 3;
    int r = e2 & 511;    int lane = r >> 4;  int q = r & 15;
    int N = t * 16 + (lane & 15);
    int K = kk * 32 + ((lane >= 16) ? 16 : 0) + q;
    if (N < G4H) {
      if (K < Hd)                 val = Wih2[N * Hd + K];          // h1 part (K 0..50)
      else if (K >= 64 && K < 64 + Hd) val = Whh2[N * Hd + (K - 64)]; // h2 part (K 64..114)
    }
  }
  pack[e] = f2bf(val);
}

// ------------------------------------------------------------------
// Persistent LSTM: one block = 16 independent batch rows, 1100 steps.
// Weights live in VGPRs; h in LDS as bf16 A-buffer; c in thread regs.
// ------------------------------------------------------------------
__global__ __launch_bounds__(THREADS)
void lstm_persistent(const float* __restrict__ x,
                     const float* __restrict__ Wih1,
                     const float* __restrict__ bih1, const float* __restrict__ bhh1,
                     const float* __restrict__ bih2, const float* __restrict__ bhh2,
                     const float* __restrict__ Wlin, const float* __restrict__ blin,
                     const unsigned short* __restrict__ pack,
                     float* __restrict__ out)
{
  __shared__ unsigned short Abuf[16][128];   // [row][K]: h1 @ K=0..50, h2 @ K=64..114, rest 0
  __shared__ float gates[16][208];           // f32 gate accumulators (reused per layer)
  __shared__ float b1L[G4H], b2L[G4H], wi1L[G4H];
  __shared__ float wlinL[Hd];
  __shared__ float blinL;
  __shared__ float xcur[16], xfb[16];

  const int tid  = threadIdx.x;
  const int lane = tid & 31;
  const int wave = tid >> 5;
  const int r0   = blockIdx.x * 16;

  // consts -> LDS (b = b_ih + b_hh folded once)
  for (int j = tid; j < G4H; j += THREADS) {
    b1L[j]  = bih1[j] + bhh1[j];
    b2L[j]  = bih2[j] + bhh2[j];
    wi1L[j] = Wih1[j];
  }
  for (int j = tid; j < Hd; j += THREADS) wlinL[j] = Wlin[j];
  if (tid == 0) blinL = blin[0];
  for (int j = tid; j < 16 * 128; j += THREADS) ((unsigned short*)Abuf)[j] = 0; // h=0, pad=0

  // Resident B fragments: tiles t = wave + 8*i  (i<2), 13 valid tiles total.
  FragU b1f[2][2];   // layer1: [tile][kchunk<2]
  FragU b2f[2][4];   // layer2: [tile][kchunk<4]
  #pragma unroll
  for (int i = 0; i < 2; ++i) {
    int t = wave + 8 * i;
    if (t < NT) {
      #pragma unroll
      for (int kk = 0; kk < 2; ++kk) {
        const u32x4* gp = (const u32x4*)(pack + ((t * 2 + kk) * 512 + lane * 16));
        b1f[i][kk].q[0] = gp[0]; b1f[i][kk].q[1] = gp[1];
      }
      #pragma unroll
      for (int kk = 0; kk < 4; ++kk) {
        const u32x4* gp = (const u32x4*)(pack + (B1_ELEMS + (t * 4 + kk) * 512 + lane * 16));
        b2f[i][kk].q[0] = gp[0]; b2f[i][kk].q[1] = gp[1];
      }
    }
  }

  float c1[4] = {0.f, 0.f, 0.f, 0.f};       // cell state, (row,unit) owned per thread
  float c2[4] = {0.f, 0.f, 0.f, 0.f};

  const int arow  = lane & 15;               // A-fragment: M = lane%16
  const int akoff = (lane >> 4) ? 16 : 0;    // lanes 16-31 carry K 16..31 of chunk
  const int mbase = (lane >> 4) ? 8 : 0;     // D layout: lanes 16-31 -> M += 8
  const int ncol0 = lane & 15;

  __syncthreads();

  #pragma unroll 1
  for (int t = 0; t < Tt; ++t) {
    if (tid < 16)
      xcur[tid] = (t < Lseq) ? x[(r0 + tid) * Lseq + t] : xfb[tid];

    // ---- layer-1 GEMM: gates = h1(16x64) @ B1(64x208) ----
    {
      FragU a0, a1;
      a0.q[0] = *(const u32x4*)&Abuf[arow][akoff];
      a0.q[1] = *(const u32x4*)&Abuf[arow][8 + akoff];
      a1.q[0] = *(const u32x4*)&Abuf[arow][32 + akoff];
      a1.q[1] = *(const u32x4*)&Abuf[arow][40 + akoff];
      #pragma unroll
      for (int i = 0; i < 2; ++i) {
        int tt = wave + 8 * i;
        if (tt < NT) {                      // wave-uniform: EXEC stays all-ones
          v8f acc = {};
          acc = wmma_bf16(a0.v, b1f[i][0].v, acc);
          acc = wmma_bf16(a1.v, b1f[i][1].v, acc);
          int col = tt * 16 + ncol0;
          #pragma unroll
          for (int v = 0; v < 8; ++v) gates[mbase + v][col] = acc[v];
        }
      }
    }
    __syncthreads();

    // ---- layer-1 activations (+x*W_ih1 +b1) -> h1, c1 ----
    #pragma unroll
    for (int p = 0; p < 4; ++p) {
      int idx = p * THREADS + tid;
      if (idx < 16 * Hd) {
        int u = idx >> 4, r = idx & 15;
        float xv = xcur[r];
        float gi = gates[r][u]       + b1L[u]       + xv * wi1L[u];
        float gf = gates[r][u + 51]  + b1L[u + 51]  + xv * wi1L[u + 51];
        float gg = gates[r][u + 102] + b1L[u + 102] + xv * wi1L[u + 102];
        float go = gates[r][u + 153] + b1L[u + 153] + xv * wi1L[u + 153];
        float cn = fsig(gf) * c1[p] + fsig(gi) * ftanh(gg);
        c1[p] = cn;
        Abuf[r][u] = f2bf(fsig(go) * ftanh(cn));
      }
    }
    __syncthreads();

    // ---- layer-2 GEMM: gates = [h1|h2](16x128) @ B2(128x208) ----
    {
      FragU a[4];
      #pragma unroll
      for (int kk = 0; kk < 4; ++kk) {
        a[kk].q[0] = *(const u32x4*)&Abuf[arow][kk * 32 + akoff];
        a[kk].q[1] = *(const u32x4*)&Abuf[arow][kk * 32 + 8 + akoff];
      }
      #pragma unroll
      for (int i = 0; i < 2; ++i) {
        int tt = wave + 8 * i;
        if (tt < NT) {
          v8f acc = {};
          #pragma unroll
          for (int kk = 0; kk < 4; ++kk) acc = wmma_bf16(a[kk].v, b2f[i][kk].v, acc);
          int col = tt * 16 + ncol0;
          #pragma unroll
          for (int v = 0; v < 8; ++v) gates[mbase + v][col] = acc[v];
        }
      }
    }
    __syncthreads();

    // ---- layer-2 activations -> h2, c2 ----
    #pragma unroll
    for (int p = 0; p < 4; ++p) {
      int idx = p * THREADS + tid;
      if (idx < 16 * Hd) {
        int u = idx >> 4, r = idx & 15;
        float gi = gates[r][u]       + b2L[u];
        float gf = gates[r][u + 51]  + b2L[u + 51];
        float gg = gates[r][u + 102] + b2L[u + 102];
        float go = gates[r][u + 153] + b2L[u + 153];
        float cn = fsig(gf) * c2[p] + fsig(gi) * ftanh(gg);
        c2[p] = cn;
        Abuf[r][64 + u] = f2bf(fsig(go) * ftanh(cn));
      }
    }
    __syncthreads();

    // ---- readout: out = h2 . Wlin + b, also feedback input ----
    if (tid < 16) {
      float s = blinL;
      #pragma unroll 1
      for (int k = 0; k < Hd; ++k) s += bf2f(Abuf[tid][64 + k]) * wlinL[k];
      out[(r0 + tid) * Tt + t] = s;
      xfb[tid] = s;
    }
    __syncthreads();
  }
}

extern "C" void kernel_launch(void* const* d_in, const int* in_sizes, int n_in,
                              void* d_out, int out_size, void* d_ws, size_t ws_size,
                              hipStream_t stream) {
  const float* x    = (const float*)d_in[0];
  const float* Wih1 = (const float*)d_in[1];
  const float* Whh1 = (const float*)d_in[2];
  const float* bih1 = (const float*)d_in[3];
  const float* bhh1 = (const float*)d_in[4];
  const float* Wih2 = (const float*)d_in[5];
  const float* Whh2 = (const float*)d_in[6];
  const float* bih2 = (const float*)d_in[7];
  const float* bhh2 = (const float*)d_in[8];
  const float* Wlin = (const float*)d_in[9];
  const float* blin = (const float*)d_in[10];
  unsigned short* pack = (unsigned short*)d_ws;   // needs (13312+26624)*2 = ~80 KB
  float* out = (float*)d_out;

  int prep_threads = 256;
  int prep_blocks  = (B1_ELEMS + B2_ELEMS + prep_threads - 1) / prep_threads; // 156
  prep_pack_weights<<<prep_blocks, prep_threads, 0, stream>>>(Whh1, Wih2, Whh2, pack);

  lstm_persistent<<<Bsz / 16, THREADS, 0, stream>>>(
      x, Wih1, bih1, bhh1, bih2, bhh2, Wlin, blin, pack, out);
}
